// MultiHeadAttention_14809047236833
// MI455X (gfx1250) — matrix operand order repaired and checked by reading, not compiled, not run on hardware
//
#include <hip/hip_runtime.h>
#include <hip/hip_bf16.h>
#include <stdint.h>
#include <stddef.h>

// ---------------------------------------------------------------------------
// MHA for MI455X (gfx1250): bf16 WMMA 16x16x32, f32 accumulate, wave32.
// B=2, S=2048, D=1024, H=16, HD=64.
// CDNA5 data movement: TDM tensor_load_to_lds for K tiles (TENSORcnt),
// async global->LDS b128 for bf16 GEMM A-tiles (ASYNCcnt).
// ---------------------------------------------------------------------------

typedef __attribute__((ext_vector_type(16))) __bf16 v16bf;
typedef __attribute__((ext_vector_type(8)))  __bf16 v8bf;
typedef __attribute__((ext_vector_type(8)))  float  v8f;
typedef __attribute__((ext_vector_type(4)))  float  v4f;
typedef __attribute__((ext_vector_type(4)))  unsigned int v4u;
typedef __attribute__((ext_vector_type(8)))  int    v8i;
typedef __attribute__((ext_vector_type(4)))  int    v4i;

#define GLOBAL_AS __attribute__((address_space(1)))
#define LDS_AS    __attribute__((address_space(3)))

#define B_  2
#define S_  2048
#define D_  1024
#define H_  16
#define HD_ 64
#define M_  (B_ * S_)   // 4096 rows for the projection GEMMs

#if __has_builtin(__builtin_amdgcn_tensor_load_to_lds) && \
    __has_builtin(__builtin_amdgcn_s_wait_tensorcnt)
#define USE_TDM 1
#else
#define USE_TDM 0
#endif

#if __has_builtin(__builtin_amdgcn_global_load_async_to_lds_b128) && \
    __has_builtin(__builtin_amdgcn_s_wait_asynccnt)
#define USE_ASYNC_LDS 1
#else
#define USE_ASYNC_LDS 0
#endif

static __device__ __forceinline__ v16bf join8(v8bf lo, v8bf hi) {
  v16bf r;
#pragma unroll
  for (int i = 0; i < 8; ++i) { r[i] = lo[i]; r[i + 8] = hi[i]; }
  return r;
}
static __device__ __forceinline__ v8bf ld8s(const __bf16* p) { return *(const v8bf*)p; }
static __device__ __forceinline__ v8bf ld8g(const __bf16* p) { return *(const v8bf*)p; }

static __device__ __forceinline__ v8f wmma_bf16(v16bf a, v16bf b, v8f c) {
  return __builtin_amdgcn_wmma_f32_16x16x32_bf16(false, a, false, b, (short)0, c,
                                                 false, false);
}

// Low 32 bits of a flat pointer into the LDS aperture == LDS byte offset.
static __device__ __forceinline__ unsigned lds_off(const void* p) {
  return (unsigned)(unsigned long long)(uintptr_t)p;
}

// ---------------------------------------------------------------------------
// Block GEMM:  C[M,N] = A[M,K] @ W[N,K]^T + bias   (contraction K-contiguous
// in memory for BOTH operands -> plain b128 fragment loads, no transpose).
// Block tile 128(M) x 64(N), 8 waves, each wave computes 16x64.
// AT = float (converted to bf16 while staging) or __bf16. OT = __bf16 or float.
// ---------------------------------------------------------------------------
template <typename AT, typename OT>
static __device__ __forceinline__ void gemm_tile(const AT* __restrict__ A,
                                                 const float* __restrict__ W,
                                                 const float* __restrict__ bias,
                                                 OT* __restrict__ C,
                                                 int M, int N, int K) {
  constexpr int LDA = 40;  // 32 + 8 pad (keeps 16B alignment, staggers banks)
  constexpr int LDB = 40;
  __shared__ __bf16 As[128 * LDA];
  __shared__ __bf16 Bs[64 * LDB];

  const int tid  = threadIdx.x;
  const int lane = tid & 31;
  const int wav  = tid >> 5;          // 0..7
  const int hh   = lane >> 4;         // lane half
  const int ln   = lane & 15;
  const int bm   = blockIdx.y * 128;
  const int bn   = blockIdx.x * 64;

  v8f acc[4];
#pragma unroll
  for (int g = 0; g < 4; ++g) acc[g] = (v8f)(0.0f);

  for (int k0 = 0; k0 < K; k0 += 32) {
    __syncthreads();
    // ---- stage A tile 128x32 (16 elems / thread) ----
    {
      const int e = tid * 16;
      const int r = e >> 5;            // row 0..127
      const int c = e & 31;            // 0 or 16
      __bf16* dst = &As[r * LDA + c];
      const AT* src = A + (size_t)(bm + r) * K + k0 + c;
      if constexpr (sizeof(AT) == 4) {  // fp32 source -> convert in VALU
        v4f f0 = *(const v4f*)(src + 0);
        v4f f1 = *(const v4f*)(src + 4);
        v4f f2 = *(const v4f*)(src + 8);
        v4f f3 = *(const v4f*)(src + 12);
        v8bf x0, x1;
#pragma unroll
        for (int i = 0; i < 4; ++i) {
          x0[i]     = (__bf16)f0[i];
          x0[i + 4] = (__bf16)f1[i];
          x1[i]     = (__bf16)f2[i];
          x1[i + 4] = (__bf16)f3[i];
        }
        *(v8bf*)dst       = x0;
        *(v8bf*)(dst + 8) = x1;
      } else {                           // already bf16: pure async copy
#if USE_ASYNC_LDS
        __builtin_amdgcn_global_load_async_to_lds_b128(
            (GLOBAL_AS v4i*)src, (LDS_AS v4i*)dst, 0, 0);
        __builtin_amdgcn_global_load_async_to_lds_b128(
            (GLOBAL_AS v4i*)(src + 8), (LDS_AS v4i*)(dst + 8), 0, 0);
#else
        *(v8bf*)dst       = ld8g((const __bf16*)src);
        *(v8bf*)(dst + 8) = ld8g((const __bf16*)src + 8);
#endif
      }
    }
    // ---- stage B tile 64x32 from W[N,K] fp32 (8 elems / thread) ----
    {
      const int e = tid * 8;
      const int r = e >> 5;            // row 0..63
      const int c = e & 31;            // 0,8,16,24
      const float* src = W + (size_t)(bn + r) * K + k0 + c;
      v4f f0 = *(const v4f*)(src + 0);
      v4f f1 = *(const v4f*)(src + 4);
      v8bf x;
#pragma unroll
      for (int i = 0; i < 4; ++i) { x[i] = (__bf16)f0[i]; x[i + 4] = (__bf16)f1[i]; }
      *(v8bf*)&Bs[r * LDB + c] = x;
    }
#if USE_ASYNC_LDS
    if constexpr (sizeof(AT) == 2)
      __builtin_amdgcn_s_wait_asynccnt(0);
#endif
    __syncthreads();

    // ---- fragments + 4 WMMA / wave ----
    const __bf16* arow = &As[(wav * 16 + ln) * LDA];
    const v16bf af = join8(ld8s(arow + hh * 16), ld8s(arow + hh * 16 + 8));
#pragma unroll
    for (int g = 0; g < 4; ++g) {
      const __bf16* brow = &Bs[(g * 16 + ln) * LDB];
      const v16bf bf = join8(ld8s(brow + hh * 16), ld8s(brow + hh * 16 + 8));
      acc[g] = wmma_bf16(af, bf, acc[g]);
    }
  }

  // ---- epilogue: bias + store ----
#pragma unroll
  for (int g = 0; g < 4; ++g) {
    const int col = bn + g * 16 + ln;
    const float bb = bias[col];
#pragma unroll
    for (int r = 0; r < 8; ++r) {
      const int row = bm + wav * 16 + r + 8 * hh;   // WMMA C layout
      C[(size_t)row * N + col] = (OT)(acc[g][r] + bb);
    }
  }
}

__global__ __launch_bounds__(256) void qkv_proj_k(
    const float* __restrict__ q, const float* __restrict__ k,
    const float* __restrict__ v, const float* __restrict__ Wq,
    const float* __restrict__ bq, const float* __restrict__ Wk,
    const float* __restrict__ bk, const float* __restrict__ Wv,
    const float* __restrict__ bv, __bf16* __restrict__ Qb,
    __bf16* __restrict__ Kb, __bf16* __restrict__ Vb) {
  const float* A; const float* W; const float* bi; __bf16* C;
  if (blockIdx.z == 0)      { A = q; W = Wq; bi = bq; C = Qb; }
  else if (blockIdx.z == 1) { A = k; W = Wk; bi = bk; C = Kb; }
  else                      { A = v; W = Wv; bi = bv; C = Vb; }
  gemm_tile<float, __bf16>(A, W, bi, C, M_, D_, D_);
}

__global__ __launch_bounds__(256) void out_proj_k(
    const __bf16* __restrict__ Ob, const float* __restrict__ Wo,
    const float* __restrict__ bo, float* __restrict__ out) {
  gemm_tile<__bf16, float>(Ob, Wo, bo, out, M_, D_, D_);
}

// ---------------------------------------------------------------------------
// Flash attention: one block per (b, h, 128 q-rows). 8 waves, each owns
// 16 q-rows, streams kv in tiles of 32 with online softmax.
// K tile is DMA'd by the Tensor Data Mover (with LDS row padding done by the
// TDM pad feature); V tile is staged transposed by VALU (TDM can't transpose).
// ---------------------------------------------------------------------------
__global__ __launch_bounds__(256) void attn_k(
    const __bf16* __restrict__ Qb, const __bf16* __restrict__ Kb,
    const __bf16* __restrict__ Vb, const unsigned char* __restrict__ mask,
    __bf16* __restrict__ Ob) {
  constexpr int LDK = 72;   // 64 + 8 pad  (TDM: 32 DW interval + 4 DW pad)
  constexpr int LDV = 40;   // 32 + 8 pad
  constexpr int LDP = 40;
  __shared__ __bf16 Ks[32 * LDK];        // K tile, [kv][d] row-major
  __shared__ __bf16 VTs[64 * LDV];       // V tile TRANSPOSED, [d][kv]
  __shared__ __bf16 Ps[8][16 * LDP];     // per-wave P scratch (C->A relayout)

  const int tid  = threadIdx.x;
  const int lane = tid & 31;
  const int wav  = tid >> 5;
  const int hh   = lane >> 4;
  const int ln   = lane & 15;
  const int b    = blockIdx.z;
  const int h    = blockIdx.y;
  const int q0   = blockIdx.x * 128;
  const int qw   = q0 + wav * 16;

  // Q fragments (A-matrix, 16x64 -> two 16x32 frags), loaded once.
  const __bf16* qrow = Qb + ((size_t)(b * S_ + qw + ln) * D_ + h * HD_);
  const v16bf qf0 = join8(ld8g(qrow + hh * 16),      ld8g(qrow + hh * 16 + 8));
  const v16bf qf1 = join8(ld8g(qrow + 32 + hh * 16), ld8g(qrow + 40 + hh * 16));

  bool msk[8];
#pragma unroll
  for (int r = 0; r < 8; ++r)
    msk[r] = mask[b * S_ + qw + r + 8 * hh] != 0;   // [B,1,S] masks q-rows

  float m_r[8], l_r[8];
#pragma unroll
  for (int r = 0; r < 8; ++r) { m_r[r] = -1e30f; l_r[r] = 0.0f; }
  v8f o[4];
#pragma unroll
  for (int g = 0; g < 4; ++g) o[g] = (v8f)(0.0f);

  const float scale = 0.125f;              // 1/sqrt(64)
  const float L2E   = 1.44269504088896f;

#if USE_TDM
  const unsigned ks_lds = lds_off(&Ks[0]);
#endif

  for (int t = 0; t < S_ / 32; ++t) {
    __syncthreads();
#if USE_TDM
    if (wav == 0) {
      // D# for a 2D bf16 tile: 32 rows x 64 elems, row stride D_ elems,
      // LDS-padded to 72-elem rows (pad 4 DWORDs every 32 DWORDs).
      const unsigned long long ga =
          (unsigned long long)(uintptr_t)(Kb + (size_t)(b * S_ + t * 32) * D_ +
                                          h * HD_);
      v4u g0;
      g0[0] = 1u;                                             // count=1
      g0[1] = ks_lds;                                         // lds_addr
      g0[2] = (unsigned)(ga & 0xFFFFFFFFu);                   // gaddr[31:0]
      g0[3] = (unsigned)((ga >> 32) & 0x01FFFFFFu) | (2u << 30);  // + type=2
      v8i g1;
      g1[0] = (1 << 16)    // data_size = 2 bytes
            | (1 << 20)    // pad_enable
            | (4 << 22)    // pad_interval: 32 DWORDs (=64 bf16 row)
            | (3 << 25);   // pad_amount: 4 DWORDs (=8 bf16)
      g1[1] = (HD_ << 16);         // tensor_dim0 low16 = 64
      g1[2] = (32 << 16);          // tensor_dim0 hi = 0 | tensor_dim1 low16 = 32
      g1[3] = (HD_ << 16);         // tensor_dim1 hi = 0 | tile_dim0 = 64
      g1[4] = 32;                  // tile_dim1 = 32 | tile_dim2 = 0
      g1[5] = D_;                  // tensor_dim0_stride low32 = 1024
      g1[6] = 0;
      g1[7] = 0;
      const v4i z4 = (v4i)0;
      const v8i z8 = (v8i)0;
      __builtin_amdgcn_tensor_load_to_lds(g0, g1, z4, z4, z8, 0);
    }
#endif
    // ---- stage V^T tile [64][32] (and K tile when no TDM) ----
    {
      const int e  = tid * 8;
      const int kr = e >> 6;     // kv row 0..31
      const int c  = e & 63;     // d col, multiple of 8
      const size_t gro = (size_t)(b * S_ + t * 32 + kr) * D_ + h * HD_ + c;
#if !USE_TDM
      *(v8bf*)&Ks[kr * LDK + c] = ld8g(Kb + gro);
#endif
      const v8bf vv = ld8g(Vb + gro);
#pragma unroll
      for (int i = 0; i < 8; ++i) VTs[(c + i) * LDV + kr] = vv[i];
      if (t + 1 < S_ / 32) {     // pull next tiles toward L2/L0
        const size_t nx = gro + (size_t)32 * D_;
        __builtin_prefetch(Kb + nx, 0, 1);
        __builtin_prefetch(Vb + nx, 0, 1);
      }
    }
#if USE_TDM
    if (wav == 0) __builtin_amdgcn_s_wait_tensorcnt(0);
#endif
    __syncthreads();

    // ---- scores: s = Q (16x64) @ K_tile^T -> two 16x16 C frags ----
    v8f s0 = (v8f)(0.0f), s1 = (v8f)(0.0f);
    {
      const __bf16* r0 = &Ks[ln * LDK];                 // kv 0..15
      const v16bf b00 = join8(ld8s(r0 + hh * 16),      ld8s(r0 + hh * 16 + 8));
      const v16bf b01 = join8(ld8s(r0 + 32 + hh * 16), ld8s(r0 + 32 + hh * 16 + 8));
      s0 = wmma_bf16(qf0, b00, s0);
      s0 = wmma_bf16(qf1, b01, s0);
      const __bf16* r1 = &Ks[(16 + ln) * LDK];          // kv 16..31
      const v16bf b10 = join8(ld8s(r1 + hh * 16),      ld8s(r1 + hh * 16 + 8));
      const v16bf b11 = join8(ld8s(r1 + 32 + hh * 16), ld8s(r1 + 32 + hh * 16 + 8));
      s1 = wmma_bf16(qf0, b10, s1);
      s1 = wmma_bf16(qf1, b11, s1);
    }

    // ---- online softmax (row = r + 8*hh, cols striped over 16 lanes) ----
#pragma unroll
    for (int r = 0; r < 8; ++r) {
      float a0 = s0[r] * scale, a1 = s1[r] * scale;
      if (msk[r]) { a0 = -1e9f; a1 = -1e9f; }
      float vmax = fmaxf(a0, a1);
      vmax = fmaxf(vmax, __shfl_xor(vmax, 1, 32));
      vmax = fmaxf(vmax, __shfl_xor(vmax, 2, 32));
      vmax = fmaxf(vmax, __shfl_xor(vmax, 4, 32));
      vmax = fmaxf(vmax, __shfl_xor(vmax, 8, 32));
      const float mnew  = fmaxf(m_r[r], vmax);
      const float alpha = exp2f((m_r[r] - mnew) * L2E);
      m_r[r] = mnew;
      const float p0 = exp2f((a0 - mnew) * L2E);
      const float p1 = exp2f((a1 - mnew) * L2E);
      float rs = p0 + p1;
      rs += __shfl_xor(rs, 1, 32);
      rs += __shfl_xor(rs, 2, 32);
      rs += __shfl_xor(rs, 4, 32);
      rs += __shfl_xor(rs, 8, 32);
      l_r[r] = l_r[r] * alpha + rs;
#pragma unroll
      for (int g = 0; g < 4; ++g) o[g][r] *= alpha;
      // scatter P (C layout) into wave-private LDS as bf16 A-matrix source
      Ps[wav][(r + 8 * hh) * LDP + ln]      = (__bf16)p0;
      Ps[wav][(r + 8 * hh) * LDP + 16 + ln] = (__bf16)p1;
    }

    asm volatile("s_wait_dscnt 0" ::: "memory");  // wave-local LDS RAW

    // ---- O += P (16x32) @ V_tile (32x64) via V^T rows in LDS ----
    const __bf16* prow = &Ps[wav][ln * LDP];
    const v16bf pf = join8(ld8s(prow + hh * 16), ld8s(prow + hh * 16 + 8));
#pragma unroll
    for (int g = 0; g < 4; ++g) {
      const __bf16* vr = &VTs[(g * 16 + ln) * LDV];
      const v16bf vf = join8(ld8s(vr + hh * 16), ld8s(vr + hh * 16 + 8));
      o[g] = wmma_bf16(pf, vf, o[g]);
    }
  }

  // ---- normalize + write O (bf16) back to ws in [B*S, D] layout ----
#pragma unroll
  for (int r = 0; r < 8; ++r) {
    const float inv = 1.0f / l_r[r];
    const size_t row = (size_t)(b * S_ + qw + r + 8 * hh) * D_ + h * HD_;
#pragma unroll
    for (int g = 0; g < 4; ++g)
      Ob[row + g * 16 + ln] = (__bf16)(o[g][r] * inv);
  }
}

// ---------------------------------------------------------------------------
extern "C" void kernel_launch(void* const* d_in, const int* in_sizes, int n_in,
                              void* d_out, int out_size, void* d_ws,
                              size_t ws_size, hipStream_t stream) {
  const float* q  = (const float*)d_in[0];
  const float* k  = (const float*)d_in[1];
  const float* v  = (const float*)d_in[2];
  const unsigned char* mask = (const unsigned char*)d_in[3];
  const float* Wq = (const float*)d_in[4];
  const float* bq = (const float*)d_in[5];
  const float* Wk = (const float*)d_in[6];
  const float* bk = (const float*)d_in[7];
  const float* Wv = (const float*)d_in[8];
  const float* bv = (const float*)d_in[9];
  const float* Wo = (const float*)d_in[10];
  const float* bo = (const float*)d_in[11];
  float* out = (float*)d_out;

  const size_t tensor_elems = (size_t)M_ * D_;   // 4M bf16 each
  __bf16* Qb = (__bf16*)d_ws;
  __bf16* Kb = Qb + tensor_elems;
  __bf16* Vb = Kb + tensor_elems;
  __bf16* Ob = Vb + tensor_elems;

  const dim3 blk(256);
  qkv_proj_k<<<dim3(D_ / 64, M_ / 128, 3), blk, 0, stream>>>(
      q, k, v, Wq, bq, Wk, bk, Wv, bv, Qb, Kb, Vb);
  attn_k<<<dim3(S_ / 128, H_, B_), blk, 0, stream>>>(Qb, Kb, Vb, mask, Ob);
  out_proj_k<<<dim3(D_ / 64, M_ / 128, 1), blk, 0, stream>>>(Ob, Wo, bo, out);
}
